// ODERNN_20933670601176
// MI455X (gfx1250) — matrix operand order repaired and checked by reading, not compiled, not run on hardware
//
#include <hip/hip_runtime.h>
#include <hip/hip_bf16.h>

// Problem constants (from reference)
#define BB   32
#define SS   128
#define NN   34          // P*J
#define DIN  2
#define DOUT 3
#define DRNN 128
#define DHID 256
#define DTC  0.1f

typedef __attribute__((ext_vector_type(16))) __bf16 v16bf;
typedef __attribute__((ext_vector_type(8)))  __bf16 v8bf;
typedef __attribute__((ext_vector_type(8)))  float  v8f;

__device__ __forceinline__ float sigmoidf_(float x) { return 1.0f / (1.0f + __expf(-x)); }

__device__ __forceinline__ v8f wmma_bf16(v16bf a, v16bf b, v8f c) {
  return __builtin_amdgcn_wmma_f32_16x16x32_bf16(false, a, false, b, (short)0, c, false, false);
}

// A fragment (16x32 bf16) from bf16 LDS, contiguous k order (k = 16*hi + s, matching packed B):
// lane's 16 K-elements are two adjacent 16B runs -> 2x ds_load_b128, no cvt.
__device__ __forceinline__ v16bf load_a_frag_bf(const __bf16* A, int lda, int kt, int lane) {
  const int m  = lane & 15;
  const int hi = lane >> 4;
  const __bf16* base = A + m * lda + (kt << 5) + (hi << 4);
  v8bf g0 = *(const v8bf*)(base);
  v8bf g1 = *(const v8bf*)(base + 8);
  v16bf a;
#pragma unroll
  for (int s = 0; s < 8; ++s) { a[s] = g0[s]; a[8 + s] = g1[s]; }
  return a;
}

// NT n-tiles (tile-index stride NST) sharing one A fragment per k-step.
template <int NT, int NKT, int NST>
__device__ __forceinline__ void gemm_bf(const __bf16* A, int lda, const __bf16* Bp,
                                        int nt0, v8f* acc, int lane) {
  const v16bf* B = (const v16bf*)Bp;
#pragma unroll
  for (int kt = 0; kt < NKT; ++kt) {
    v16bf a = load_a_frag_bf(A, lda, kt, lane);
#pragma unroll
    for (int j = 0; j < NT; ++j)
      acc[j] = wmma_bf16(a, B[((nt0 + j * NST) * NKT + kt) * 32 + lane], acc[j]);
  }
}

// Pack a row-major f32 weight [K, Ncols] into bf16 WMMA B-fragment layout.
__global__ void pack_w_kernel(const float* __restrict__ src, __bf16* __restrict__ dst,
                              int K, int Ncols) {
  int idx = blockIdx.x * blockDim.x + threadIdx.x;
  int total = (K >> 5) * (Ncols >> 4) * 512;
  if (idx >= total) return;
  int slot = idx & 15;
  int lane = (idx >> 4) & 31;
  int tile = idx >> 9;
  int KT = K >> 5;
  int nt = tile / KT;
  int kt = tile - nt * KT;
  int col = (nt << 4) + (lane & 15);
  int k = (kt << 5) + slot + ((lane >> 4) << 4);
  dst[idx] = (__bf16)src[k * Ncols + col];
}

// Pack wout (128x3) zero-padded to one 128x16 B tile (4 k-fragments).
__global__ void pack_wout_kernel(const float* __restrict__ src, __bf16* __restrict__ dst) {
  int idx = blockIdx.x * blockDim.x + threadIdx.x;
  if (idx >= 2048) return;
  int slot = idx & 15;
  int lane = (idx >> 4) & 31;
  int kt = idx >> 9;
  int col = lane & 15;
  int k = (kt << 5) + slot + ((lane >> 4) << 4);
  dst[idx] = (col < DOUT) ? (__bf16)src[k * DOUT + col] : (__bf16)(0.0f);
}

__global__ __launch_bounds__(256) void odernn_kernel(
    const float* __restrict__ x2d, const int* __restrict__ mask,
    const float* __restrict__ ode_b0, const float* __restrict__ ode_b1,
    const float* __restrict__ ode_b2,
    const float* __restrict__ wih0, const float* __restrict__ bih0,
    const float* __restrict__ bhh0,
    const float* __restrict__ bih1, const float* __restrict__ bhh1,
    const float* __restrict__ bout,
    const float* __restrict__ h0,
    const __bf16* __restrict__ p_w0, const __bf16* __restrict__ p_w1,
    const __bf16* __restrict__ p_w2, const __bf16* __restrict__ p_whh0,
    const __bf16* __restrict__ p_wih1, const __bf16* __restrict__ p_whh1,
    const __bf16* __restrict__ p_wout,
    float* __restrict__ out) {
  __shared__ float  sh_h [16 * DRNN];                   //  8 KB f32 state h
  __shared__ __bf16 sh_hb[16 * DRNN];                   //  4 KB bf16 shadow of h (GEMM A)
  __shared__ __bf16 sh_h2b[16 * DRNN];                  //  4 KB bf16 GRU0 output (GEMM A)
  __shared__ __align__(16) unsigned char sh_buf0[16 * 384 * 4];  // 24 KB
  __shared__ __align__(16) unsigned char sh_buf1[16 * 384 * 4];  // 24 KB

  __bf16* t0b = (__bf16*)sh_buf0;   // ODE a1, 16x256 bf16
  __bf16* t1b = (__bf16*)sh_buf1;   // ODE a2, 16x256 bf16
  float*  g0f = (float*)sh_buf1;    // GRU0 gh, 16x384 f32
  float*  gif = (float*)sh_buf0;    // GRU1 gi, 16x384 f32
  float*  ghf = (float*)sh_buf1;    // GRU1 gh, 16x384 f32

  const int tid  = threadIdx.x;
  const int lane = tid & 31;
  const int wave = tid >> 5;
  const int m0   = blockIdx.x << 4;
  const int colc = lane & 15;
  const int hi   = lane >> 4;

  // Per-thread loop-invariant gate-phase indices: thread handles (r0+2k, cc).
  const int r0 = tid >> 7;
  const int cc = tid & 127;
  int cbase[8];
#pragma unroll
  for (int k = 0; k < 8; ++k) {
    int g = m0 + r0 + 2 * k;
    int b = g / NN, n = g - b * NN;
    cbase[k] = b * SS * NN + n;       // + t*NN gives the (b,t,n) flat index
  }

  for (int i = tid; i < 16 * DRNN; i += 256) {
    float v = h0[i & (DRNN - 1)];
    sh_h[i] = v;
    sh_hb[i] = (__bf16)v;
  }
  __syncthreads();

#pragma unroll 1
  for (int t = 0; t < SS; ++t) {
    // ---- 4 Euler ODE steps ----
#pragma unroll 1
    for (int e = 0; e < 4; ++e) {
      {                                                  // L1: N=256, K=128
        v8f acc[2] = {v8f{}, v8f{}};
        gemm_bf<2, 4, 8>(sh_hb, DRNN, p_w0, wave, acc, lane);
#pragma unroll
        for (int j = 0; j < 2; ++j) {
          int col = ((wave + (j << 3)) << 4) + colc;
          float bv = ode_b0[col];
#pragma unroll
          for (int v = 0; v < 8; ++v)
            t0b[(v + (hi << 3)) * DHID + col] = (__bf16)tanhf(acc[j][v] + bv);
        }
      }
      __syncthreads();
      {                                                  // L2: N=256, K=256
        v8f acc[2] = {v8f{}, v8f{}};
        gemm_bf<2, 8, 8>(t0b, DHID, p_w1, wave, acc, lane);
#pragma unroll
        for (int j = 0; j < 2; ++j) {
          int col = ((wave + (j << 3)) << 4) + colc;
          float bv = ode_b1[col];
#pragma unroll
          for (int v = 0; v < 8; ++v)
            t1b[(v + (hi << 3)) * DHID + col] = (__bf16)tanhf(acc[j][v] + bv);
        }
      }
      __syncthreads();
      {                                                  // L3: N=128, K=256, split-K x2
        const v16bf* B = (const v16bf*)p_w2;
        v8f a0 = {}, a1 = {};
#pragma unroll
        for (int kt = 0; kt < 4; ++kt) {
          v16bf x0 = load_a_frag_bf(t1b, DHID, 2 * kt,     lane);
          v16bf x1 = load_a_frag_bf(t1b, DHID, 2 * kt + 1, lane);
          v16bf b0 = B[(wave * 8 + 2 * kt)     * 32 + lane];
          v16bf b1 = B[(wave * 8 + 2 * kt + 1) * 32 + lane];
          a0 = wmma_bf16(x0, b0, a0);
          a1 = wmma_bf16(x1, b1, a1);
        }
        int col = (wave << 4) + colc;
        float bv = ode_b2[col];
#pragma unroll
        for (int v = 0; v < 8; ++v) {
          int row = v + (hi << 3);
          float hn = sh_h[row * DRNN + col] + DTC * (a0[v] + a1[v] + bv);
          sh_h[row * DRNN + col] = hn;
          sh_hb[row * DRNN + col] = (__bf16)hn;
        }
      }
      __syncthreads();
    }

    // ---- y_t = h1 @ wout + bout as one padded WMMA tile on wave 0 ----
    if (wave == 0) {
      const v16bf* B = (const v16bf*)p_wout;
      v8f acc = {};
#pragma unroll
      for (int kt = 0; kt < 4; ++kt)
        acc = wmma_bf16(load_a_frag_bf(sh_hb, DRNN, kt, lane), B[kt * 32 + lane], acc);
      if (colc < DOUT) {                 // lanes 0-2 (rows 0-7) and 16-18 (rows 8-15)
        float bv = bout[colc];
#pragma unroll
        for (int v = 0; v < 8; ++v) {
          int row = v + (hi << 3);
          int g = m0 + row, b = g / NN, n = g - b * NN;
          out[((b * SS + t) * NN + n) * DOUT + colc] = acc[v] + bv;
        }
      }
    }

    // ---- GRU0: gh = h1 @ whh0 + bhh0 (N=384 -> 3 tiles per wave) ----
    {
      v8f acc[3] = {v8f{}, v8f{}, v8f{}};
      gemm_bf<3, 4, 8>(sh_hb, DRNN, p_whh0, wave, acc, lane);
#pragma unroll
      for (int j = 0; j < 3; ++j) {
        int col = ((wave + (j << 3)) << 4) + colc;
        float bv = bhh0[col];
#pragma unroll
        for (int v = 0; v < 8; ++v)
          g0f[(v + (hi << 3)) * 384 + col] = acc[j][v] + bv;
      }
    }
    __syncthreads();

    // GRU0 gates (gi: K=2 on VALU) -> h2 (bf16, GEMM A operand only)
#pragma unroll
    for (int k = 0; k < 8; ++k) {
      int r = r0 + 2 * k;
      int off = cbase[k] + t * NN;
      float mm = (float)mask[off];
      float x0 = x2d[2 * off] * mm, x1 = x2d[2 * off + 1] * mm;
      float ir = x0 * wih0[cc]       + x1 * wih0[384 + cc]       + bih0[cc];
      float iz = x0 * wih0[128 + cc] + x1 * wih0[384 + 128 + cc] + bih0[128 + cc];
      float ig = x0 * wih0[256 + cc] + x1 * wih0[384 + 256 + cc] + bih0[256 + cc];
      float hr = g0f[r * 384 + cc], hz = g0f[r * 384 + 128 + cc], hg = g0f[r * 384 + 256 + cc];
      float rr = sigmoidf_(ir + hr), zz = sigmoidf_(iz + hz);
      float nn_ = tanhf(ig + rr * hg);
      float h1v = sh_h[r * DRNN + cc];
      sh_h2b[r * DRNN + cc] = (__bf16)((1.0f - zz) * nn_ + zz * h1v);
    }
    __syncthreads();

    // ---- GRU1: gi = h2 @ wih1 + bih1 ; gh = h1 @ whh1 + bhh1 ----
    {
      v8f acc[3] = {v8f{}, v8f{}, v8f{}};
      gemm_bf<3, 4, 8>(sh_h2b, DRNN, p_wih1, wave, acc, lane);
#pragma unroll
      for (int j = 0; j < 3; ++j) {
        int col = ((wave + (j << 3)) << 4) + colc;
        float bv = bih1[col];
#pragma unroll
        for (int v = 0; v < 8; ++v)
          gif[(v + (hi << 3)) * 384 + col] = acc[j][v] + bv;
      }
    }
    {
      v8f acc[3] = {v8f{}, v8f{}, v8f{}};
      gemm_bf<3, 4, 8>(sh_hb, DRNN, p_whh1, wave, acc, lane);
#pragma unroll
      for (int j = 0; j < 3; ++j) {
        int col = ((wave + (j << 3)) << 4) + colc;
        float bv = bhh1[col];
#pragma unroll
        for (int v = 0; v < 8; ++v)
          ghf[(v + (hi << 3)) * 384 + col] = acc[j][v] + bv;
      }
    }
    __syncthreads();

    // GRU1 gates + mask blend -> next h (f32 + bf16 shadow)
#pragma unroll
    for (int k = 0; k < 8; ++k) {
      int r = r0 + 2 * k;
      int off = cbase[k] + t * NN;
      float mm = (float)mask[off];
      float ir = gif[r * 384 + cc], iz = gif[r * 384 + 128 + cc], ig = gif[r * 384 + 256 + cc];
      float hr = ghf[r * 384 + cc], hz = ghf[r * 384 + 128 + cc], hg = ghf[r * 384 + 256 + cc];
      float rr = sigmoidf_(ir + hr), zz = sigmoidf_(iz + hz);
      float nn_ = tanhf(ig + rr * hg);
      float h1v = sh_h[r * DRNN + cc];
      float h2v = (1.0f - zz) * nn_ + zz * h1v;
      float hn = h1v * (1.0f - mm) + h2v * mm;
      sh_h[r * DRNN + cc] = hn;
      sh_hb[r * DRNN + cc] = (__bf16)hn;
    }
    __syncthreads();
  }
}

extern "C" void kernel_launch(void* const* d_in, const int* in_sizes, int n_in,
                              void* d_out, int out_size, void* d_ws, size_t ws_size,
                              hipStream_t stream) {
  const float* x2d    = (const float*)d_in[0];
  const int*   mask   = (const int*)  d_in[1];
  const float* ode_w0 = (const float*)d_in[2];
  const float* ode_b0 = (const float*)d_in[3];
  const float* ode_w1 = (const float*)d_in[4];
  const float* ode_b1 = (const float*)d_in[5];
  const float* ode_w2 = (const float*)d_in[6];
  const float* ode_b2 = (const float*)d_in[7];
  const float* wih0   = (const float*)d_in[8];
  const float* whh0   = (const float*)d_in[9];
  const float* bih0   = (const float*)d_in[10];
  const float* bhh0   = (const float*)d_in[11];
  const float* wih1   = (const float*)d_in[12];
  const float* whh1   = (const float*)d_in[13];
  const float* bih1   = (const float*)d_in[14];
  const float* bhh1   = (const float*)d_in[15];
  const float* wout   = (const float*)d_in[16];
  const float* bout   = (const float*)d_in[17];
  const float* h0     = (const float*)d_in[18];

  // bf16-packed weights in workspace (561,152 bytes total)
  __bf16* p_w0   = (__bf16*)d_ws;       // 128x256 -> 32768 elems
  __bf16* p_w1   = p_w0 + 32768;        // 256x256 -> 65536
  __bf16* p_w2   = p_w1 + 65536;        // 256x128 -> 32768
  __bf16* p_whh0 = p_w2 + 32768;        // 128x384 -> 49152
  __bf16* p_wih1 = p_whh0 + 49152;      // 128x384 -> 49152
  __bf16* p_whh1 = p_wih1 + 49152;      // 128x384 -> 49152
  __bf16* p_wout = p_whh1 + 49152;      // 128x16 (padded) -> 2048

  pack_w_kernel<<<(32768 + 255) / 256, 256, 0, stream>>>(ode_w0, p_w0, 128, 256);
  pack_w_kernel<<<(65536 + 255) / 256, 256, 0, stream>>>(ode_w1, p_w1, 256, 256);
  pack_w_kernel<<<(32768 + 255) / 256, 256, 0, stream>>>(ode_w2, p_w2, 256, 128);
  pack_w_kernel<<<(49152 + 255) / 256, 256, 0, stream>>>(whh0, p_whh0, 128, 384);
  pack_w_kernel<<<(49152 + 255) / 256, 256, 0, stream>>>(wih1, p_wih1, 128, 384);
  pack_w_kernel<<<(49152 + 255) / 256, 256, 0, stream>>>(whh1, p_whh1, 128, 384);
  pack_wout_kernel<<<8, 256, 0, stream>>>(wout, p_wout);

  // 1088 rows / 16-row WMMA tiles = 68 persistent workgroups, 8 wave32s each
  odernn_kernel<<<68, 256, 0, stream>>>(x2d, mask, ode_b0, ode_b1, ode_b2,
                                        wih0, bih0, bhh0, bih1, bhh1,
                                        bout, h0,
                                        p_w0, p_w1, p_w2, p_whh0, p_wih1, p_whh1, p_wout,
                                        (float*)d_out);
  (void)in_sizes; (void)n_in; (void)out_size; (void)ws_size;
}
